// YOLOLayer_77721728188790
// MI455X (gfx1250) — matrix operand order repaired and checked by reading, not compiled, not run on hardware
//
#include <hip/hip_runtime.h>
#include <hip/hip_bf16.h>

typedef __attribute__((ext_vector_type(16))) __bf16 v16bf;
typedef __attribute__((ext_vector_type(8)))  float  v8f;

#define F_    13
#define P_    169     // 13*13
#define PPAD  192     // padded p (6 blocks * 32)
#define O_    255
#define OPAD  256
#define K_    1024
#define NB    16
#define NGT   800
#define LPB   50

// ---------- helpers ----------
__device__ __forceinline__ unsigned short f32_bf16_rne(float f) {
  unsigned u = __float_as_uint(f);
  unsigned r = 0x7FFFu + ((u >> 16) & 1u);
  return (unsigned short)((u + r) >> 16);
}
__device__ __forceinline__ float bf16_f32(unsigned short h) {
  return __uint_as_float(((unsigned)h) << 16);
}
__device__ __forceinline__ void bf16_split(float f, unsigned short* hi, unsigned short* lo) {
  unsigned short h = f32_bf16_rne(f);
  *hi = h;
  *lo = f32_bf16_rne(f - bf16_f32(h));
}
__device__ __forceinline__ float sigm(float x) { return 1.f / (1.f + __expf(-x)); }
__device__ __forceinline__ float bce(float p, float t) {
  p = fminf(fmaxf(p, 1e-12f), 1.f - 1e-12f);
  return -(t * __logf(p) + (1.f - t) * __logf(1.f - p));
}

// ---------- prep: W (255x1024 f32) -> bf16 hi/lo, padded to 256 rows ----------
__global__ __launch_bounds__(256)
void conv_w(const float* __restrict__ W, unsigned short* __restrict__ whi,
            unsigned short* __restrict__ wlo) {
  const int idx = blockIdx.x * 256 + threadIdx.x;     // 256*1024 total
  const int o = idx >> 10, k = idx & 1023;
  const float v = (o < O_) ? W[(size_t)o * K_ + k] : 0.f;
  bf16_split(v, &whi[idx], &wlo[idx]);
}

// ---------- prep: X (n,k,p) f32 -> bf16 hi/lo transposed to (n,p,k), p padded ----------
// LDS-tiled 32x32 transpose: coalesced reads (p-contiguous) and writes (k-contiguous).
__global__ __launch_bounds__(256)
void conv_x_t(const float* __restrict__ X, unsigned short* __restrict__ xhi,
              unsigned short* __restrict__ xlo) {
  __shared__ float tile[32][33];
  const int n  = blockIdx.z;
  const int kb = blockIdx.y * 32;
  const int pb = blockIdx.x * 32;
  const int t  = threadIdx.x;
  {
    const int lp = t & 31, lk0 = t >> 5;
    #pragma unroll
    for (int s = 0; s < 4; ++s) {
      const int lk = lk0 + 8 * s;
      const int p = pb + lp;
      tile[lk][lp] = (p < P_) ? X[((size_t)n * K_ + kb + lk) * P_ + p] : 0.f;
    }
  }
  __syncthreads();
  {
    const int lk = t & 31, lp0 = t >> 5;
    #pragma unroll
    for (int s = 0; s < 4; ++s) {
      const int lp = lp0 + 8 * s;
      const size_t o = ((size_t)n * PPAD + pb + lp) * K_ + kb + lk;
      bf16_split(tile[lk][lp], &xhi[o], &xlo[o]);
    }
  }
}

// ---------- GEMM: out[n,o,p] = sum_k x[n,k,p]*W[o,k] + b[o] ----------
// 256 threads = 8 wave32; block tile 64(o) x 32(p); bf16 hi/lo -> 3 WMMA per K-step.
// Tiles streamed global->LDS with CDNA5 async loads, double-buffered.
__global__ __launch_bounds__(256)
void gemm_wmma(const unsigned short* __restrict__ Whi, const unsigned short* __restrict__ Wlo,
               const unsigned short* __restrict__ Xhi, const unsigned short* __restrict__ Xlo,
               const float* __restrict__ Bias, float* __restrict__ out) {
  __shared__ unsigned short Ah[2][64][32];   // [buf][o_local][k_local]
  __shared__ unsigned short Al[2][64][32];
  __shared__ unsigned short Bh[2][32][32];   // [buf][p_local][k_local]
  __shared__ unsigned short Bl[2][32][32];

  const int tid    = threadIdx.x;
  const int n      = blockIdx.z;
  const int o_base = blockIdx.y * 64;
  const int p_base = blockIdx.x * 32;

  // loader mapping (no guards: operands are zero-padded)
  const int ow = tid >> 2;          // 0..63
  const int kw = (tid & 3) << 3;    // 8 bf16 = 16B per thread (b128)
  const int px = tid >> 3;          // 0..31
  const int kx = (tid & 7) << 2;    // 4 bf16 = 8B per thread (b64)

  const unsigned short* wh = Whi + (size_t)(o_base + ow) * K_ + kw;
  const unsigned short* wl = Wlo + (size_t)(o_base + ow) * K_ + kw;
  const unsigned short* xh = Xhi + ((size_t)n * PPAD + p_base + px) * K_ + kx;
  const unsigned short* xl = Xlo + ((size_t)n * PPAD + p_base + px) * K_ + kx;

  unsigned ldsAh[2], ldsAl[2], ldsBh[2], ldsBl[2];
  #pragma unroll
  for (int b2 = 0; b2 < 2; ++b2) {
    ldsAh[b2] = (unsigned)(uintptr_t)&Ah[b2][ow][kw];
    ldsAl[b2] = (unsigned)(uintptr_t)&Al[b2][ow][kw];
    ldsBh[b2] = (unsigned)(uintptr_t)&Bh[b2][px][kx];
    ldsBl[b2] = (unsigned)(uintptr_t)&Bl[b2][px][kx];
  }

#define ISSUE_TILE(buf, koff)                                                                  \
  do {                                                                                         \
    asm volatile("global_load_async_to_lds_b128 %0, %1, off"                                   \
                 :: "v"(ldsAh[buf]), "v"(wh + (koff)) : "memory");                             \
    asm volatile("global_load_async_to_lds_b128 %0, %1, off"                                   \
                 :: "v"(ldsAl[buf]), "v"(wl + (koff)) : "memory");                             \
    asm volatile("global_load_async_to_lds_b64 %0, %1, off"                                    \
                 :: "v"(ldsBh[buf]), "v"(xh + (koff)) : "memory");                             \
    asm volatile("global_load_async_to_lds_b64 %0, %1, off"                                    \
                 :: "v"(ldsBl[buf]), "v"(xl + (koff)) : "memory");                             \
  } while (0)

  v8f acc = {};
  const int lane  = tid & 31;
  const int wave  = tid >> 5;
  const int wo    = wave & 3;
  const int wp    = wave >> 2;
  const int m     = lane & 15;
  const int kaddA = (lane < 16) ? 0 : 8;    // A frag: half-wave K split (ISA 7.12.2)
  const int kbB   = (lane < 16) ? 0 : 16;   // unused: B frag uses kaddA split

  ISSUE_TILE(0, 0);                         // prologue: stage tile 0

  const int NIT = K_ / 32;
  for (int it = 0; it < NIT; ++it) {
    const int cur = it & 1;
    if (it + 1 < NIT) {
      ISSUE_TILE(1 - cur, (it + 1) * 32);                   // stream next tile
      asm volatile("s_wait_asynccnt 0x4" ::: "memory");     // oldest 4 (cur tile) done
    } else {
      asm volatile("s_wait_asynccnt 0x0" ::: "memory");
    }
    __syncthreads();

    union Frag { unsigned int u[8]; v16bf v; } fah, fal, fbh, fbl;
    #pragma unroll
    for (int j = 0; j < 8; ++j) {
      const int ka = (j < 4) ? (2 * j + kaddA) : (16 + 2 * (j - 4) + kaddA);
      fah.u[j] = *(const unsigned int*)&Ah[cur][wo * 16 + m][ka];
      fal.u[j] = *(const unsigned int*)&Al[cur][wo * 16 + m][ka];
      const int kb = (j < 4) ? (2 * j + kaddA) : (16 + 2 * (j - 4) + kaddA);
      fbh.u[j] = *(const unsigned int*)&Bh[cur][wp * 16 + m][kb];
      fbl.u[j] = *(const unsigned int*)&Bl[cur][wp * 16 + m][kb];
    }
    acc = __builtin_amdgcn_wmma_f32_16x16x32_bf16(false, fah.v, false, fbh.v, (short)0, acc, false, false);
    acc = __builtin_amdgcn_wmma_f32_16x16x32_bf16(false, fah.v, false, fbl.v, (short)0, acc, false, false);
    acc = __builtin_amdgcn_wmma_f32_16x16x32_bf16(false, fal.v, false, fbh.v, (short)0, acc, false, false);
    __syncthreads();
  }
#undef ISSUE_TILE

  // D layout: VGPR r -> M = r + (lane>=16 ? 8 : 0), N = lane&15
  const int col = lane & 15;
  const int p_g = p_base + wp * 16 + col;
  const int mo  = (lane < 16) ? 0 : 8;
  if (p_g < P_) {
    #pragma unroll
    for (int r = 0; r < 8; ++r) {
      const int o_g = o_base + wo * 16 + mo + r;
      if (o_g < O_)
        out[((size_t)n * O_ + o_g) * P_ + p_g] = acc[r] + Bias[o_g];
    }
  }
}

// ---------- per-GT target precompute ----------
__global__ void gt_prep(const float* __restrict__ labels, float4* __restrict__ gtbox,
                        float* tx, float* ty, float* tw, float* th, float* tsc,
                        int* key, int* bm, int* cls) {
  int g = blockIdx.x * blockDim.x + threadIdx.x;
  if (g >= NGT) return;
  const float ALLW[9] = {10.f/32, 16.f/32, 33.f/32, 30.f/32, 62.f/32, 59.f/32, 116.f/32, 156.f/32, 373.f/32};
  const float ALLH[9] = {13.f/32, 30.f/32, 23.f/32, 61.f/32, 45.f/32, 119.f/32, 90.f/32, 198.f/32, 326.f/32};
  const float* lab = labels + (size_t)g * 5;
  const int n = g / LPB;
  const float gx = lab[1] * F_, gy = lab[2] * F_, gw = lab[3] * F_, gh = lab[4] * F_;
  const int gl = (int)lab[0];
  float best = -1e30f; int bn = 0;
  #pragma unroll
  for (int k = 0; k < 9; ++k) {
    float ai = fminf(gw, ALLW[k]) * fminf(gh, ALLH[k]);
    float au = gw * gh + ALLW[k] * ALLH[k] - ai;
    float r  = ai / au;
    if (r > best) { best = r; bn = k; }   // argmax: first max wins
  }
  const int a  = bn % 3;
  const int gi = (int)gx, gj = (int)gy;
  gtbox[g] = make_float4(gx, gy, gw, gh);
  tx[g]  = gx - floorf(gx);
  ty[g]  = gy - floorf(gy);
  tw[g]  = __logf(gw / ALLW[6 + a] + 1e-16f);
  th[g]  = __logf(gh / ALLH[6 + a] + 1e-16f);
  tsc[g] = sqrtf(2.f - gw * gh / (float)P_);
  key[g] = (n * 3 + a) * P_ + gj * F_ + gi;   // flat cell id
  bm[g]  = (bn >= 6) ? 1 : 0;
  cls[g] = gl;
}

// ---------- loss: one thread per (n,a,y,x) cell ----------
__global__ __launch_bounds__(256)
void yolo_loss(const float* __restrict__ pred, const float4* __restrict__ gtbox,
               const float* __restrict__ tx, const float* __restrict__ ty,
               const float* __restrict__ tw, const float* __restrict__ th,
               const float* __restrict__ tsc, const int* __restrict__ key,
               const int* __restrict__ bm, const int* __restrict__ cls,
               float* __restrict__ out) {
  __shared__ float4 sbox[NGT];
  __shared__ int    skey[NGT];
  __shared__ int    sbm[NGT];
  const int tid = threadIdx.x;
  for (int g = tid; g < NGT; g += 256) {
    sbox[g] = gtbox[g]; skey[g] = key[g]; sbm[g] = bm[g];
  }
  __syncthreads();

  const int cell = blockIdx.x * 256 + tid;
  float lx = 0, ly = 0, lw = 0, lh = 0, lobj = 0, lcls = 0;

  if (cell < NB * 3 * P_) {
    const int n   = cell / (3 * P_);
    const int rem = cell % (3 * P_);
    const int a   = rem / P_;
    const int p   = rem % P_;
    const int jj  = p / F_, ii = p % F_;
    const float MW[3] = {116.f/32, 156.f/32, 373.f/32};
    const float MH[3] = {90.f/32, 198.f/32, 326.f/32};

    const float* cb = pred + ((size_t)n * O_ + a * 85) * P_ + p;
    const float pcx  = sigm(cb[0 * P_]);
    const float pcy  = sigm(cb[1 * P_]);
    const float pwr  = cb[2 * P_];
    const float phr  = cb[3 * P_];
    const float pobj = sigm(cb[4 * P_]);

    const float bx = pcx + (float)ii, by = pcy + (float)jj;
    const float bw = MW[a] * __expf(pwr), bh = MH[a] * __expf(phr);
    const float hbw = 0.5f * bw, hbh = 0.5f * bh, areab = bw * bh;

    float maxiou = -1e30f;
    int tgt = -1, objflag = 0;
    for (int g = 0; g < NGT; ++g) {
      const float4 gb = sbox[g];
      const float tlx = fmaxf(bx - hbw, gb.x - 0.5f * gb.z);
      const float tly = fmaxf(by - hbh, gb.y - 0.5f * gb.w);
      const float brx = fminf(bx + hbw, gb.x + 0.5f * gb.z);
      const float bry = fminf(by + hbh, gb.y + 0.5f * gb.w);
      const float wi = fmaxf(brx - tlx, 0.f);
      const float hi = fmaxf(bry - tly, 0.f);
      const float inter = wi * hi;
      const float iou = inter / (areab + gb.z * gb.w - inter);
      maxiou = fmaxf(maxiou, iou);
      if (skey[g] == cell) { tgt = g; objflag |= sbm[g]; }  // scatter: last gt wins; obj needs bmask
    }
    const bool obj   = (objflag != 0);
    const bool noobj = (maxiou <= 0.7f) || obj;   // noobj forced true at bmask target cells
    const float tobj = obj ? 1.f : 0.f;
    if (noobj) lobj = bce(pobj, tobj);
    if (obj) {
      const float sc = tsc[tgt];
      const float w2 = sc * sc;
      lx = w2 * bce(pcx, tx[tgt]);
      ly = w2 * bce(pcy, ty[tgt]);
      const float dw = pwr - tw[tgt];
      const float dh = phr - th[tgt];
      lw = 0.5f * w2 * dw * dw;
      lh = 0.5f * w2 * dh * dh;
      const int cl = cls[tgt];
      for (int c = 0; c < 80; ++c) {
        const float pc = sigm(cb[(5 + c) * P_]);
        lcls += bce(pc, (c == cl) ? 1.f : 0.f);
      }
    }
  }

  const float ltot = lx + ly + lw + lh + lobj + lcls;
  float vals[7] = {ltot, lx, ly, lw, lh, lobj, lcls};
  #pragma unroll
  for (int i = 0; i < 7; ++i) {
    float v = vals[i];
    for (int o = 16; o > 0; o >>= 1) v += __shfl_xor(v, o, 32);  // wave32 reduce
    if ((tid & 31) == 0) atomicAdd(&out[i], v);
  }
}

__global__ void zero_out7(float* out) { if (threadIdx.x < 7) out[threadIdx.x] = 0.f; }

extern "C" void kernel_launch(void* const* d_in, const int* in_sizes, int n_in,
                              void* d_out, int out_size, void* d_ws, size_t ws_size,
                              hipStream_t stream) {
  (void)in_sizes; (void)n_in; (void)out_size; (void)ws_size;
  const float* xin    = (const float*)d_in[0];   // (16,1024,13,13)
  const float* labels = (const float*)d_in[1];   // (16,50,5)
  const float* W      = (const float*)d_in[2];   // (255,1024)
  const float* b      = (const float*)d_in[3];   // (255,)
  float* out = (float*)d_out;                    // 7 floats

  // ---- workspace layout (all sub-offsets 16B aligned) ----
  float*  gemm_out = (float*)d_ws;                       // 16*255*169 floats
  size_t  go       = (size_t)NB * O_ * P_;
  float4* gtbox = (float4*)(gemm_out + go);
  float*  tx  = (float*)(gtbox + NGT);
  float*  ty  = tx + NGT;
  float*  tw  = ty + NGT;
  float*  th  = tw + NGT;
  float*  tsc = th + NGT;
  int*    key = (int*)(tsc + NGT);
  int*    bm  = key + NGT;
  int*    cls = bm + NGT;
  unsigned short* whi = (unsigned short*)(cls + NGT);    // 256*1024
  unsigned short* wlo = whi + (size_t)OPAD * K_;
  unsigned short* xhi = wlo + (size_t)OPAD * K_;         // 16*192*1024
  unsigned short* xlo = xhi + (size_t)NB * PPAD * K_;

  zero_out7<<<1, 32, 0, stream>>>(out);
  gt_prep<<<dim3((NGT + 255) / 256), 256, 0, stream>>>(labels, gtbox, tx, ty, tw, th, tsc, key, bm, cls);
  conv_w<<<dim3((OPAD * K_) / 256), 256, 0, stream>>>(W, whi, wlo);
  conv_x_t<<<dim3(PPAD / 32, K_ / 32, NB), 256, 0, stream>>>(xin, xhi, xlo);
  gemm_wmma<<<dim3(6, 4, 16), 256, 0, stream>>>(whi, wlo, xhi, xlo, b, gemm_out);
  yolo_loss<<<dim3(32), 256, 0, stream>>>(gemm_out, gtbox, tx, ty, tw, th, tsc, key, bm, cls, out);
}
